// CausalSelfAttention_28217935135250
// MI455X (gfx1250) — matrix operand order repaired and checked by reading, not compiled, not run on hardware
//
#include <hip/hip_runtime.h>
#include <hip/hip_bf16.h>

typedef __attribute__((ext_vector_type(16))) __bf16 v16bf;
typedef __attribute__((ext_vector_type(8)))  float  v8f;
typedef __attribute__((ext_vector_type(4)))  unsigned int v4u;
typedef __attribute__((ext_vector_type(8)))  int    v8i;
typedef __attribute__((ext_vector_type(4)))  int    v4i;
typedef unsigned short u16;
typedef unsigned int   u32;
typedef unsigned long long u64;

#define DIM  1536
#define NH   12
#define HD   128
#define S    2640       // 22*40*3
#define PAST 10560      // 12*22*40
#define KV   13200      // PAST + S
#define HW_  880        // H*W = 22*40
#define QSCALE 0.08838834764831845f   // 1/sqrt(128)

__device__ __forceinline__ u16 f2bf(float f) {
  u32 u = __float_as_uint(f);
  u32 r = (u + 0x7FFFu + ((u >> 16) & 1u)) >> 16;
  return (u16)r;
}

// ---------------------------------------------------------------------------
// f32 -> bf16 elementwise convert
// ---------------------------------------------------------------------------
__global__ __launch_bounds__(256) void cvt_kernel(const float* __restrict__ in,
                                                  u16* __restrict__ out, int n) {
  int i = blockIdx.x * 256 + threadIdx.x;
  if (i < n) out[i] = f2bf(in[i]);
}

// ---------------------------------------------------------------------------
// GEMM: out_row = rowwise_epilogue( A[16 x 1536] @ W^T + b )
// Block: 16 rows of A, ALL 1536 output cols (so RMSNorm is block-local).
// 256 threads = 8 waves; wave w owns N-tiles [w*12, w*12+12).
// mode 0: Q (rmsnorm*gq, rope, *QSCALE, bf16)   mode 1: K (rmsnorm*gk, rope, bf16)
// mode 2: V (bias only, bf16)                   mode 3: O-proj (bias only, f32 d_out)
// ---------------------------------------------------------------------------
__global__ __launch_bounds__(256) void gemm_kernel(
    const float* __restrict__ A, const u16* __restrict__ Wbf,
    const float* __restrict__ bias, const float* __restrict__ g,
    const float* __restrict__ freqs, const int* __restrict__ cs_ptr,
    u16* __restrict__ out_bf, float* __restrict__ out_f32, int mode) {
  extern __shared__ char smem[];
  u16*   Ax  = (u16*)smem;                       // 16*1536 bf16 = 48KB
  float* OUT = (float*)(smem + 49152);           // 16*1536 f32  = 96KB
  float* CS  = (float*)(smem + 49152 + 98304);   // 16*64*2 f32  = 8KB
  float* RED = (float*)(smem + 49152 + 98304 + 8192); // 16*16 f32

  const int tid = threadIdx.x;
  const int s0  = blockIdx.x * 16;

  // Stage A tile (f32 -> bf16) into LDS
  {
    const float4* Ap = (const float4*)(A + (size_t)s0 * DIM);
    uint2* Axp = (uint2*)Ax;
    for (int i = tid; i < (16 * DIM) / 4; i += 256) {
      float4 v = Ap[i];
      uint2 p;
      p.x = (u32)f2bf(v.x) | ((u32)f2bf(v.y) << 16);
      p.y = (u32)f2bf(v.z) | ((u32)f2bf(v.w) << 16);
      Axp[i] = p;
    }
  }
  // Precompute RoPE cos/sin for the 16 rows (modes 0/1)
  if (mode < 2) {
    int cstart = *cs_ptr;
    for (int e = tid; e < 16 * 64; e += 256) {
      int r = e >> 6, j = e & 63;
      int pos = cstart + s0 + r;
      int t = pos / HW_; int rem = pos - t * HW_;
      int hh = rem / 40; int ww = rem - hh * 40;
      float ang;
      if (j < 22)      ang = freqs[t * 64 + j];
      else if (j < 43) ang = freqs[hh * 64 + j];
      else             ang = freqs[ww * 64 + j];
      CS[e * 2 + 0] = __cosf(ang);
      CS[e * 2 + 1] = __sinf(ang);
    }
  }
  __syncthreads();

  const int w = tid >> 5, l = tid & 31;
  const int m = l & 15, half = l >> 4;

  // One-shot prefetch of this wave's W panel (global_prefetch_b8), outside
  // the hot loop so it can't perturb the WMMA software pipeline.
#pragma unroll
  for (int nt = 0; nt < 12; nt++) {
    const u16* base = Wbf + (size_t)((w * 12 + nt) * 16 + m) * DIM;
    __builtin_prefetch(base, 0, 1);
    __builtin_prefetch(base + 512, 0, 1);
    __builtin_prefetch(base + 1024, 0, 1);
  }

  v8f acc[12] = {};
  for (int kb = 0; kb < DIM; kb += 32) {
    // A fragment (16x32 bf16): lane m=l%16; K = half*8+j (j<8) / 16+half*8+(j-8)
    v16bf a;
    uint4* ap = (uint4*)&a;
    ap[0] = *(const uint4*)(Ax + m * DIM + kb + half * 8);
    ap[1] = *(const uint4*)(Ax + m * DIM + kb + 16 + half * 8);
#pragma unroll
    for (int nt = 0; nt < 12; nt++) {
      int n = (w * 12 + nt) * 16 + m;  // B lane: N = l%16
      // B fragment (32x16): K = half*8+j (j<8) / 16+half*8+(j-8) of W row n
      v16bf b;
      uint4* bp = (uint4*)&b;
      const uint4* src = (const uint4*)(Wbf + (size_t)n * DIM + kb + half * 8);
      bp[0] = src[0];
      bp[1] = src[2];
      acc[nt] = __builtin_amdgcn_wmma_f32_16x16x32_bf16(
          false, a, false, b, (short)0, acc[nt], false, false);
    }
  }

  // D layout: lane l, vgpr r -> (row = r + 8*half, col = l%16). Add bias, to LDS.
#pragma unroll
  for (int nt = 0; nt < 12; nt++) {
    int ncol = (w * 12 + nt) * 16 + m;
    float bv = bias[ncol];
#pragma unroll
    for (int r = 0; r < 8; r++)
      OUT[(r + 8 * half) * DIM + ncol] = acc[nt][r] + bv;
  }
  __syncthreads();

  const int rr = tid >> 4, lr = tid & 15;
  if (mode < 2) {
    float ss = 0.f;
    for (int i = 0; i < 96; i++) {
      float v = OUT[rr * DIM + lr + 16 * i];
      ss += v * v;
    }
    RED[rr * 16 + lr] = ss;
    __syncthreads();
    if (tid < 16) {
      float s = 0.f;
      for (int i = 0; i < 16; i++) s += RED[tid * 16 + i];
      RED[tid * 16] = rsqrtf(s * (1.0f / DIM) + 1e-6f);
    }
    __syncthreads();
    float rrms = RED[rr * 16];
    float sc = (mode == 0) ? QSCALE : 1.0f;
    u32* outp = (u32*)(out_bf + (size_t)(s0 + rr) * DIM);
    for (int i = 0; i < 48; i++) {
      int p = lr * 48 + i;
      int c0 = 2 * p;
      float e = OUT[rr * DIM + c0] * rrms * g[c0];
      float o = OUT[rr * DIM + c0 + 1] * rrms * g[c0 + 1];
      int j = p & 63;  // pair index within head
      float cv = CS[(rr * 64 + j) * 2 + 0];
      float sv = CS[(rr * 64 + j) * 2 + 1];
      float oe = (e * cv - o * sv) * sc;
      float oo = (o * cv + e * sv) * sc;
      outp[p] = (u32)f2bf(oe) | ((u32)f2bf(oo) << 16);
    }
  } else if (mode == 2) {
    u32* outp = (u32*)(out_bf + (size_t)(s0 + rr) * DIM);
    for (int i = 0; i < 48; i++) {
      int p = lr * 48 + i;
      outp[p] = (u32)f2bf(OUT[rr * DIM + 2 * p]) |
                ((u32)f2bf(OUT[rr * DIM + 2 * p + 1]) << 16);
    }
  } else {
    float* outp = out_f32 + (size_t)(s0 + rr) * DIM;
    for (int i = 0; i < 96; i++) {
      int c = lr + 16 * i;
      outp[c] = OUT[rr * DIM + c];
    }
  }
}

// ---------------------------------------------------------------------------
// Flash attention: block = 8 waves; wave owns 16 Q rows.
// K chunk (32 keys x 128 d, bf16) staged by the Tensor Data Mover
// (tensor_load_to_lds, TENSORcnt); V chunk staged transposed by the waves.
// Online softmax, f32 accumulators. grid = (ceil(165/8)=21, NH)
// ---------------------------------------------------------------------------
__global__ __launch_bounds__(256) void attn_kernel(
    const u16* __restrict__ qbf, const u16* __restrict__ kall,
    const u16* __restrict__ vall, float* __restrict__ otmp) {
  __shared__ __attribute__((aligned(16))) u16 Kl[32 * 128];   // [key][d]  (TDM dest)
  __shared__ __attribute__((aligned(16))) u16 Vt[128 * 32];   // [d][key]
  __shared__ __attribute__((aligned(16))) u16 Pl[8 * 16 * 32];// per-wave P

  const int tid = threadIdx.x;
  const int w = tid >> 5, l = tid & 31;
  const int m = l & 15, half = l >> 4;
  const int h = blockIdx.y;
  const int qt = blockIdx.x * 8 + w;
  const int qtc = (qt < 165) ? qt : 164;

  // Preload Q fragments (16x128, 4 fragments of 16x32)
  v16bf qf[4];
  const u16* qbase = qbf + (size_t)(qtc * 16 + m) * DIM + h * HD;
#pragma unroll
  for (int f = 0; f < 4; f++) {
    uint4* qp = (uint4*)&qf[f];
    qp[0] = *(const uint4*)(qbase + f * 32 + half * 8);
    qp[1] = *(const uint4*)(qbase + f * 32 + 16 + half * 8);
  }

  v8f O[8] = {};
  float runM[8], runL[8];
#pragma unroll
  for (int r = 0; r < 8; r++) { runM[r] = -1e30f; runL[r] = 0.f; }

  // Invariant D# pieces for the K-tile TDM load (ISA cdna5 ch.8 bitfields)
  const u32 ldsK   = (u32)(size_t)Kl;                       // LDS byte offset
  const u64 gbaseK = (u64)(size_t)kall + (u64)h * HD * 2;   // head base (bytes)

  const int skey = tid >> 3, spart = tid & 7;  // V staging: key 0..31, 16-elem part

  for (int kc = 0; kc < KV; kc += 32) {
    __syncthreads();  // previous chunk fully consumed

    // --- K tile via Tensor Data Mover: 2D tensor, rows=remaining keys,
    //     row len 128 elems (2B), row stride 1536 elems; tile 128x32 -> Kl.
    if (w == 0) {
      u64 ga = gbaseK + (u64)kc * (DIM * 2);
      int rem = KV - kc;  // rows past this read back zero (OOB policy)
      v4u g0 = { 1u,                                  // count=1 (valid D#)
                 ldsK,                                // lds_addr
                 (u32)ga,                             // global_addr[31:0]
                 (u32)(ga >> 32) | (2u << 30) };      // global_addr[56:32] | type=2
      v8i g1 = { (int)(1u << 16),                     // data_size=1 (2 bytes)
                 (int)(128u << 16),                   // tensor_dim0 = 128
                 (int)(((u32)rem & 0xFFFFu) << 16),   // tensor_dim1 = remaining keys
                 (int)(128u << 16),                   // tile_dim0 = 128
                 32,                                  // tile_dim1 = 32
                 DIM,                                 // tensor_dim0_stride = 1536
                 0, 0 };
      v4i gz4 = { 0, 0, 0, 0 };
      v8i gz8 = { 0, 0, 0, 0, 0, 0, 0, 0 };
      __builtin_amdgcn_tensor_load_to_lds(g0, g1, gz4, gz4, gz8, 0);
    }

    // --- V tile staged transposed by all 256 threads
    {
      int gk = kc + skey;
      uint4 z4 = {0, 0, 0, 0};
      uint4 v0 = z4, v1 = z4;
      if (gk < KV) {
        const uint4* vs = (const uint4*)(vall + (size_t)gk * DIM + h * HD + spart * 16);
        v0 = vs[0]; v1 = vs[1];
      }
      union { uint4 q[2]; u16 s[16]; } vu;
      vu.q[0] = v0; vu.q[1] = v1;
#pragma unroll
      for (int e = 0; e < 16; e++)
        Vt[(spart * 16 + e) * 32 + skey] = vu.s[e];
    }

    if (w == 0) __builtin_amdgcn_s_wait_tensorcnt(0);  // TDM done before barrier
    __syncthreads();

    // Scores: two 16x16 key tiles, K-dim = 128 in 4 steps
    v8f sa0 = {}, sa1 = {};
#pragma unroll
    for (int f = 0; f < 4; f++) {
      v16bf b0, b1;
      uint4* bp0 = (uint4*)&b0;
      uint4* bp1 = (uint4*)&b1;
      const u16* kr0 = Kl + m * 128 + f * 32 + half * 8;
      const u16* kr1 = Kl + (16 + m) * 128 + f * 32 + half * 8;
      bp0[0] = *(const uint4*)(kr0); bp0[1] = *(const uint4*)(kr0 + 16);
      bp1[0] = *(const uint4*)(kr1); bp1[1] = *(const uint4*)(kr1 + 16);
      sa0 = __builtin_amdgcn_wmma_f32_16x16x32_bf16(false, qf[f], false, b0,
                                                    (short)0, sa0, false, false);
      sa1 = __builtin_amdgcn_wmma_f32_16x16x32_bf16(false, qf[f], false, b1,
                                                    (short)0, sa1, false, false);
    }
    // Mask out-of-range keys (lane's key col = kc + tile*16 + l%16)
    if (kc + m >= KV) {
#pragma unroll
      for (int r = 0; r < 8; r++) sa0[r] = -1e30f;
    }
    if (kc + 16 + m >= KV) {
#pragma unroll
      for (int r = 0; r < 8; r++) sa1[r] = -1e30f;
    }

    // Online softmax; row values live across 16 lanes of the same half
    float alpha[8];
#pragma unroll
    for (int r = 0; r < 8; r++) {
      float mx = fmaxf(sa0[r], sa1[r]);
      for (int off = 1; off < 16; off <<= 1)
        mx = fmaxf(mx, __shfl_xor(mx, off, 32));
      float nm = fmaxf(runM[r], mx);
      alpha[r] = __expf(runM[r] - nm);
      runM[r] = nm;
      float p0 = __expf(sa0[r] - nm);
      float p1 = __expf(sa1[r] - nm);
      sa0[r] = p0; sa1[r] = p1;
      float rs = p0 + p1;
      for (int off = 1; off < 16; off <<= 1)
        rs += __shfl_xor(rs, off, 32);
      runL[r] = alpha[r] * runL[r] + rs;
    }
#pragma unroll
    for (int nt = 0; nt < 8; nt++)
#pragma unroll
      for (int r = 0; r < 8; r++) O[nt][r] *= alpha[r];

    // D-layout P -> per-wave LDS -> A-layout fragment (16x32 bf16)
    u16* pb = Pl + w * 512;
#pragma unroll
    for (int r = 0; r < 8; r++) {
      int row = r + 8 * half;
      pb[row * 32 + m]      = f2bf(sa0[r]);
      pb[row * 32 + 16 + m] = f2bf(sa1[r]);
    }
    v16bf pf;
    uint4* pp = (uint4*)&pf;
    pp[0] = *(const uint4*)(pb + m * 32 + half * 8);
    pp[1] = *(const uint4*)(pb + m * 32 + 16 + half * 8);

    // P @ V: 8 N-tiles over head dim
#pragma unroll
    for (int nt = 0; nt < 8; nt++) {
      v16bf vf;
      uint4* vp = (uint4*)&vf;
      const u16* vr = Vt + (nt * 16 + m) * 32 + half * 8;
      vp[0] = *(const uint4*)(vr);
      vp[1] = *(const uint4*)(vr + 16);
      O[nt] = __builtin_amdgcn_wmma_f32_16x16x32_bf16(false, pf, false, vf,
                                                      (short)0, O[nt], false, false);
    }
  }

  if (qt < 165) {
    float inv[8];
#pragma unroll
    for (int r = 0; r < 8; r++) inv[r] = 1.0f / runL[r];
#pragma unroll
    for (int nt = 0; nt < 8; nt++)
#pragma unroll
      for (int r = 0; r < 8; r++) {
        int row = qt * 16 + r + 8 * half;
        otmp[(size_t)row * DIM + h * HD + nt * 16 + m] = O[nt][r] * inv[r];
      }
  }
}

// ---------------------------------------------------------------------------
extern "C" void kernel_launch(void* const* d_in, const int* in_sizes, int n_in,
                              void* d_out, int out_size, void* d_ws, size_t ws_size,
                              hipStream_t stream) {
  const float* x   = (const float*)d_in[0];
  const float* fr  = (const float*)d_in[1];
  const float* ck  = (const float*)d_in[2];
  const float* cv  = (const float*)d_in[3];
  const float* Wq  = (const float*)d_in[4];
  const float* bq  = (const float*)d_in[5];
  const float* Wk  = (const float*)d_in[6];
  const float* bk  = (const float*)d_in[7];
  const float* Wv  = (const float*)d_in[8];
  const float* bv  = (const float*)d_in[9];
  const float* Wo  = (const float*)d_in[10];
  const float* bo  = (const float*)d_in[11];
  const float* gq  = (const float*)d_in[12];
  const float* gk  = (const float*)d_in[13];
  const int*   cs  = (const int*)d_in[14];

  char* ws = (char*)d_ws;
  u16*   qbf  = (u16*)(ws);                          // S*DIM bf16
  u16*   kall = (u16*)(ws + 8110080);                // KV*DIM bf16
  u16*   vall = (u16*)(ws + 48660480);               // KV*DIM bf16
  float* otmp = (float*)(ws + 89210880);             // S*DIM f32
  u16*   wqb  = (u16*)(ws + 105431040);              // DIM*DIM bf16 x4
  u16*   wkb  = (u16*)(ws + 110149632);
  u16*   wvb  = (u16*)(ws + 114868224);
  u16*   wob  = (u16*)(ws + 119586816);

  const int NW = DIM * DIM;            // 2359296
  const int NC = PAST * DIM;           // 16220160
  cvt_kernel<<<(NW + 255) / 256, 256, 0, stream>>>(Wq, wqb, NW);
  cvt_kernel<<<(NW + 255) / 256, 256, 0, stream>>>(Wk, wkb, NW);
  cvt_kernel<<<(NW + 255) / 256, 256, 0, stream>>>(Wv, wvb, NW);
  cvt_kernel<<<(NW + 255) / 256, 256, 0, stream>>>(Wo, wob, NW);
  cvt_kernel<<<(NC + 255) / 256, 256, 0, stream>>>(ck, kall, NC);
  cvt_kernel<<<(NC + 255) / 256, 256, 0, stream>>>(cv, vall, NC);

  const size_t SMEM = 49152 + 98304 + 8192 + 1024;  // 156,672 B (<320KB/WGP)
  gemm_kernel<<<S / 16, 256, SMEM, stream>>>(x, wqb, bq, gq, fr, cs,
                                             qbf, nullptr, 0);
  gemm_kernel<<<S / 16, 256, SMEM, stream>>>(x, wkb, bk, gk, fr, cs,
                                             kall + (size_t)PAST * DIM, nullptr, 1);
  gemm_kernel<<<S / 16, 256, SMEM, stream>>>(x, wvb, bv, gq, fr, cs,
                                             vall + (size_t)PAST * DIM, nullptr, 2);

  attn_kernel<<<dim3(21, NH), 256, 0, stream>>>(qbf, kall, vall, otmp);

  gemm_kernel<<<S / 16, 256, SMEM, stream>>>(otmp, wob, bo, gq, fr, cs,
                                             nullptr, (float*)d_out, 3);
}
